// SourceBasedTGNMemory_34565896798992
// MI455X (gfx1250) — compile-verified, matrix operand order checked by reading, not run on hardware
//
#include <hip/hip_runtime.h>
#include <hip/hip_fp16.h>

typedef _Float16 h16;
typedef __attribute__((ext_vector_type(16))) _Float16 v16h;
typedef __attribute__((ext_vector_type(8)))  _Float16 v8h;
typedef __attribute__((ext_vector_type(8)))  float    v8f;

#define LMAX    48
#define ALPHA   0.1f
#define NEVENTS 4096
#define NNODES  100000
#define MEMD    256
#define XSTR    552   // 544 used + 8 pad (1104B row, 16B-aligned stride)
#define HSTR    264   // 256 used + 8 pad (528B row, 16B-aligned stride)

// ---------------- init last_level ----------------
__global__ void k_init_ll(int* ll, int n) {
  int i = blockIdx.x * blockDim.x + threadIdx.x;
  if (i < n) ll[i] = -1;
}

// ---------------- weight swizzle: fp32 (O x K) row-major -> f16 WMMA B-fragments
// Per (n-tile of 16 outputs, k-chunk of 32): 32 lanes x 16 f16 (32B per lane, 1KB per tile-chunk).
// Lane/VGPR mapping follows the CDNA5 16-bit 16x32 fragment layout:
//   lane = n + 16*((kk>>3)&1) ; dword v = ((kk&16)>>2)+((kk&7)>>1) ; half = kk&1
__global__ void k_swizzle(const float* __restrict__ W, h16* __restrict__ dstp,
                          int O, int K, int Kp) {
  int i = blockIdx.x * blockDim.x + threadIdx.x;
  int total = O * Kp;
  if (i >= total) return;
  int o = i / Kp, k = i - o * Kp;
  int C = Kp >> 5;
  int tile = o >> 4, n = o & 15;
  int chunk = k >> 5, kk = k & 31;
  int lane = n + (((kk >> 3) & 1) << 4);
  int v = ((kk & 16) >> 2) + ((kk & 7) >> 1);
  int hi = kk & 1;
  float val = (k < K) ? W[(size_t)o * K + k] : 0.0f;
  dstp[(((size_t)(tile * C + chunk) * 32 + lane) * 16 + v * 2 + hi)] = (h16)val;
}

// ---------------- scheduler: greedy per-node leveling + counting sort ----------------
__global__ void k_schedule(const int* __restrict__ src, const int* __restrict__ dst,
                           int* __restrict__ ll, int* __restrict__ levels,
                           int* __restrict__ order, int* __restrict__ offs) {
  if (threadIdx.x != 0 || blockIdx.x != 0) return;
  int cnt[LMAX];
  for (int l = 0; l < LMAX; l++) cnt[l] = 0;
  for (int i = 0; i < NEVENTS; i++) {
    int s = src[i], d = dst[i];
    int ls = ll[s], ld = ll[d];
    int l = (ls > ld ? ls : ld) + 1;
    if (l >= LMAX) l = LMAX - 1;
    ll[s] = l; ll[d] = l;
    levels[i] = l;
    cnt[l]++;
  }
  int run = 0;
  for (int l = 0; l < LMAX; l++) { offs[l] = run; run += cnt[l]; cnt[l] = offs[l]; }
  offs[LMAX] = run;
  for (int i = 0; i < NEVENTS; i++) {
    int l = levels[i];
    order[cnt[l]++] = i;
  }
}

// A-fragment: 16 f16 of one event row for one 32-wide K chunk, per documented layout:
// lane half h picks K subranges {8h..8h+7} and {16+8h..16+8h+7} -> two contiguous v8h.
static __device__ inline v16h afrag(const h16* __restrict__ row, int koff, int h) {
  const h16* p = row + koff + 8 * h;
  v8h lo = *(const v8h*)(p);
  v8h hv = *(const v8h*)(p + 16);
  return __builtin_shufflevector(lo, hv, 0,1,2,3,4,5,6,7,8,9,10,11,12,13,14,15);
}

static __device__ inline v8f wmma16(v16h a, v16h b, v8f c) {
  return __builtin_amdgcn_wmma_f32_16x16x32_f16(false, a, false, b, (short)0, c, false, false);
}

// ---------------- level kernel: 1 block = 16 events, 256 threads = 8 waves ----------------
__global__ void __launch_bounds__(256)
k_level(int lvl, const int* __restrict__ order, const int* __restrict__ offs,
        const int* __restrict__ src, const int* __restrict__ dstn,
        const float* __restrict__ ef, const int* __restrict__ tsi,
        float* __restrict__ mem, float* __restrict__ lu,
        const h16* __restrict__ W1s, const float* __restrict__ b1,
        const h16* __restrict__ W2s, const float* __restrict__ b2,
        const h16* __restrict__ WIs, const float* __restrict__ bih,
        const h16* __restrict__ WHs, const float* __restrict__ bhh) {
  int beg = offs[lvl], end = offs[lvl + 1];
  int base = beg + blockIdx.x * 16;
  if (base >= end) return;                 // uniform per block
  int bcnt = end - base; if (bcnt > 16) bcnt = 16;

  __shared__ __align__(16) h16 Xa[16 * XSTR];    // [prev_src | prev_dst | ef | 0-pad] per event
  __shared__ __align__(16) h16 H1h[16 * HSTR];
  __shared__ __align__(16) h16 MSGh[16 * HSTR];
  __shared__ int   sEv[16], dEv[16];
  __shared__ float decS[16], decD[16];

  int tid = threadIdx.x;

  // ---- phase 1a: event metadata, decay factors, last_update writes ----
  if (tid < 16) {
    int e = tid;
    int s = 0, d = 0; float t = 0.f;
    if (e < bcnt) {
      int id = order[base + e];
      s = src[id]; d = dstn[id];
      t = (float)tsi[id];
      for (int c = 0; c < 3; c++) Xa[e * XSTR + 512 + c] = (h16)ef[(size_t)id * 3 + c];
    } else {
      for (int c = 0; c < 3; c++) Xa[e * XSTR + 512 + c] = (h16)0.f;
    }
    for (int c = 515; c < 544; c++) Xa[e * XSTR + c] = (h16)0.f;
    sEv[e] = s; dEv[e] = d;
    float dts = t - lu[s]; if (dts < 0.f) dts = 0.f;
    float dtd = t - lu[d]; if (dtd < 0.f) dtd = 0.f;
    decS[e] = __expf(-ALPHA * dts);
    decD[e] = __expf(-ALPHA * dtd);
    if (e < bcnt) { lu[s] = t; lu[d] = t; }   // same-level events touch disjoint nodes
  }
  __syncthreads();

  // ---- phase 1b: gather decayed memories into Xa (f16) ----
  {
    int e = tid >> 4, c0 = tid & 15;
    int s = sEv[e], d = dEv[e];
    float ds_ = decS[e], dd_ = decD[e];
    for (int j = 0; j < 16; j++) {
      int dim = c0 + 16 * j;
      Xa[e * XSTR + dim]        = (h16)(mem[(size_t)s * MEMD + dim] * ds_);
      Xa[e * XSTR + 256 + dim]  = (h16)(mem[(size_t)d * MEMD + dim] * dd_);
    }
  }
  __syncthreads();

  int lane = tid & 31, wid = tid >> 5;
  int mrow = lane & 15, half = lane >> 4;
  const h16* arowX = &Xa[mrow * XSTR];

  // ---- stage 1: H1 = relu(MsgIn(16x544) @ W1^T + b1) ----
  for (int nt = wid; nt < 16; nt += 8) {
    v8f acc = {};
    for (int c = 0; c < 17; c++) {
      v16h a = afrag(arowX, c * 32, half);
      v16h b = *(const v16h*)(W1s + ((size_t)(nt * 17 + c) * 32 + lane) * 16);
      acc = wmma16(a, b, acc);
    }
    int dim = nt * 16 + mrow;                 // D: lane -> N(dim), vgpr+half -> M(event)
    float bb = b1[dim];
    for (int r = 0; r < 8; r++) {
      float v = acc[r] + bb; if (v < 0.f) v = 0.f;
      H1h[(r + 8 * half) * HSTR + dim] = (h16)v;
    }
  }
  __syncthreads();

  // ---- stage 2: MSG = H1(16x256) @ W2^T + b2 ----
  {
    const h16* arowH = &H1h[mrow * HSTR];
    for (int nt = wid; nt < 16; nt += 8) {
      v8f acc = {};
      for (int c = 0; c < 8; c++) {
        v16h a = afrag(arowH, c * 32, half);
        v16h b = *(const v16h*)(W2s + ((size_t)(nt * 8 + c) * 32 + lane) * 16);
        acc = wmma16(a, b, acc);
      }
      int dim = nt * 16 + mrow;
      float bb = b2[dim];
      for (int r = 0; r < 8; r++)
        MSGh[(r + 8 * half) * HSTR + dim] = (h16)(acc[r] + bb);
    }
  }
  __syncthreads();

  // ---- stage 3: fused GRU gates + writeback. Per o-tile: 9 GEMM tiles (gi r/z/n, gh_src, gh_dst)
  {
    const h16* arowM = &MSGh[mrow * HSTR];
    const h16* arowS = &Xa[mrow * XSTR];          // prev_src cols 0..255
    const h16* arowD = &Xa[mrow * XSTR + 256];    // prev_dst cols 256..511
    for (int ot = wid; ot < 16; ot += 8) {
      v8f gi[3] = {{}, {}, {}}, gs[3] = {{}, {}, {}}, gd[3] = {{}, {}, {}};
      for (int c = 0; c < 8; c++) {
        v16h am = afrag(arowM, c * 32, half);
        v16h as = afrag(arowS, c * 32, half);
        v16h ad = afrag(arowD, c * 32, half);
        for (int g = 0; g < 3; g++) {
          int tile = ot + 16 * g;                 // w_* rows are [r;z;n] blocks of 256
          v16h bi = *(const v16h*)(WIs + ((size_t)(tile * 8 + c) * 32 + lane) * 16);
          v16h bh = *(const v16h*)(WHs + ((size_t)(tile * 8 + c) * 32 + lane) * 16);
          gi[g] = wmma16(am, bi, gi[g]);
          gs[g] = wmma16(as, bh, gs[g]);
          gd[g] = wmma16(ad, bh, gd[g]);
        }
      }
      int dim = ot * 16 + mrow;
      float bir = bih[dim], biz = bih[256 + dim], bin = bih[512 + dim];
      float bhr = bhh[dim], bhz = bhh[256 + dim], bhn = bhh[512 + dim];
      for (int r = 0; r < 8; r++) {
        int ev = r + 8 * half;
        if (ev >= bcnt) continue;
        float ir = gi[0][r] + bir, iz = gi[1][r] + biz, in_ = gi[2][r] + bin;
        {   // src update (written first)
          float hr = gs[0][r] + bhr, hz = gs[1][r] + bhz, hn = gs[2][r] + bhn;
          float rr = 1.f / (1.f + __expf(-(ir + hr)));
          float zz = 1.f / (1.f + __expf(-(iz + hz)));
          float nn = tanhf(in_ + rr * hn);
          float hp = (float)Xa[ev * XSTR + dim];
          mem[(size_t)sEv[ev] * MEMD + dim] = (1.f - zz) * nn + zz * hp;
        }
        {   // dst update second -> wins when s==d (same lane, program order)
          float hr = gd[0][r] + bhr, hz = gd[1][r] + bhz, hn = gd[2][r] + bhn;
          float rr = 1.f / (1.f + __expf(-(ir + hr)));
          float zz = 1.f / (1.f + __expf(-(iz + hz)));
          float nn = tanhf(in_ + rr * hn);
          float hp = (float)Xa[ev * XSTR + 256 + dim];
          mem[(size_t)dEv[ev] * MEMD + dim] = (1.f - zz) * nn + zz * hp;
        }
      }
    }
  }
}

extern "C" void kernel_launch(void* const* d_in, const int* in_sizes, int n_in,
                              void* d_out, int out_size, void* d_ws, size_t ws_size,
                              hipStream_t stream) {
  (void)in_sizes; (void)n_in; (void)out_size; (void)ws_size;
  const int*   src  = (const int*)d_in[0];
  const int*   dstn = (const int*)d_in[1];
  const float* ef   = (const float*)d_in[2];
  const int*   ts   = (const int*)d_in[3];
  const float* mem0 = (const float*)d_in[4];
  const float* lu0  = (const float*)d_in[5];
  const float* W1   = (const float*)d_in[6];
  const float* b1   = (const float*)d_in[7];
  const float* W2   = (const float*)d_in[8];
  const float* b2   = (const float*)d_in[9];
  const float* wih  = (const float*)d_in[10];
  const float* whh  = (const float*)d_in[11];
  const float* bih  = (const float*)d_in[12];
  const float* bhh  = (const float*)d_in[13];
  float* mem = (float*)d_out;

  char* ws = (char*)d_ws;
  auto alloc = [&](size_t bytes) -> char* {
    char* p = ws; ws += (bytes + 511) & ~((size_t)511); return p;
  };
  int*   order  = (int*)alloc((size_t)NEVENTS * 4);
  int*   levels = (int*)alloc((size_t)NEVENTS * 4);
  int*   offs   = (int*)alloc((size_t)(LMAX + 1) * 4);
  int*   ll     = (int*)alloc((size_t)NNODES * 4);
  float* lu     = (float*)alloc((size_t)NNODES * 4);
  h16*   W1s    = (h16*)alloc((size_t)16 * 17 * 32 * 16 * 2);
  h16*   W2s    = (h16*)alloc((size_t)16 * 8  * 32 * 16 * 2);
  h16*   WIs    = (h16*)alloc((size_t)48 * 8  * 32 * 16 * 2);
  h16*   WHs    = (h16*)alloc((size_t)48 * 8  * 32 * 16 * 2);

  hipMemcpyAsync(mem, mem0, (size_t)NNODES * MEMD * 4, hipMemcpyDeviceToDevice, stream);
  hipMemcpyAsync(lu,  lu0,  (size_t)NNODES * 4,        hipMemcpyDeviceToDevice, stream);

  k_init_ll<<<(NNODES + 255) / 256, 256, 0, stream>>>(ll, NNODES);
  k_swizzle<<<(256 * 544 + 255) / 256, 256, 0, stream>>>(W1,  W1s, 256, 515, 544);
  k_swizzle<<<(256 * 256 + 255) / 256, 256, 0, stream>>>(W2,  W2s, 256, 256, 256);
  k_swizzle<<<(768 * 256 + 255) / 256, 256, 0, stream>>>(wih, WIs, 768, 256, 256);
  k_swizzle<<<(768 * 256 + 255) / 256, 256, 0, stream>>>(whh, WHs, 768, 256, 256);
  k_schedule<<<1, 1, 0, stream>>>(src, dstn, ll, levels, order, offs);

  for (int l = 0; l < LMAX; l++)
    k_level<<<256, 256, 0, stream>>>(l, order, offs, src, dstn, ef, ts,
                                     mem, lu, W1s, b1, W2s, b2, WIs, bih, WHs, bhh);
}
